// CSOCRGAttention_34454227648697
// MI455X (gfx1250) — matrix operand order repaired and checked by reading, not compiled, not run on hardware
//
#include <hip/hip_runtime.h>

typedef __attribute__((ext_vector_type(16))) _Float16     v16h;
typedef __attribute__((ext_vector_type(8)))  float        v8f;
typedef __attribute__((ext_vector_type(4)))  unsigned int v4u;
typedef __attribute__((ext_vector_type(8)))  int          v8i;
typedef __attribute__((ext_vector_type(4)))  int          v4i;

#define LAMBDA_RG 64.0f
#define B_   4
#define N_   4096
#define D_   512
#define MTILE 32     // output rows per workgroup (2 row-tiles of 16)
#define KTILE 32     // K step matches v_wmma_f32_16x16x32_f16
#define THREADS 256  // 8 wave32
#define NSTEPS (N_ / KTILE)

// ---------------------------------------------------------------------------
// Pre-pass: latent[b][n][d] f32  ->  latT[b][d][n] f16  (one-time convert+transpose)
// ---------------------------------------------------------------------------
__global__ __launch_bounds__(256)
void latent_cvt_transpose(const float* __restrict__ latent,
                          _Float16* __restrict__ latT)
{
    __shared__ float tile[32][33];
    const int b  = blockIdx.z;
    const int n0 = blockIdx.x * 32;
    const int d0 = blockIdx.y * 32;
    const int tx = threadIdx.x;   // 0..31
    const int ty = threadIdx.y;   // 0..7
    const size_t inBase = (size_t)b * N_ * D_;

    #pragma unroll
    for (int yy = 0; yy < 4; ++yy) {
        const int n = ty + yy * 8;
        tile[n][tx] = latent[inBase + (size_t)(n0 + n) * D_ + d0 + tx];
    }
    __syncthreads();
    #pragma unroll
    for (int yy = 0; yy < 4; ++yy) {
        const int dd = ty + yy * 8;
        latT[(size_t)b * D_ * N_ + (size_t)(d0 + dd) * N_ + n0 + tx] =
            (_Float16)tile[tx][dd];
    }
}

// ---------------------------------------------------------------------------
// TDM: DMA one 32(k) x 512(col) f16 tile, laid out [col][k] in LDS.
// D# per CDNA5 ISA ch.8: 2D tensor, dim0 = n (contiguous, len N_), dim1 = d.
// ---------------------------------------------------------------------------
__device__ __forceinline__ void tdm_load_tile(unsigned ldsOff, const _Float16* gptr)
{
    const unsigned long long ga = (unsigned long long)(size_t)gptr;
    v4u g0;
    g0[0] = 1u;                                                // count=1, user D#
    g0[1] = ldsOff;                                            // lds_addr (bytes)
    g0[2] = (unsigned)(ga & 0xFFFFFFFFu);                      // global_addr[31:0]
    g0[3] = (unsigned)((ga >> 32) & 0x1FFFFFFu) | (2u << 30);  // addr[56:32] | type=2
    v8i g1;
    g1[0] = (int)(1u << 16);                                   // data_size=1 (2 bytes)
    g1[1] = (int)((unsigned)(N_ & 0xFFFF) << 16);              // tensor_dim0[15:0]
    g1[2] = (int)(((unsigned)N_ >> 16) |
                  ((unsigned)(D_ & 0xFFFF) << 16));            // dim0 hi | tensor_dim1 lo
    g1[3] = (int)(((unsigned)D_ >> 16) |
                  ((unsigned)KTILE << 16));                    // dim1 hi | tile_dim0=32
    g1[4] = (int)((unsigned)D_);                               // tile_dim1=512, tile_dim2=0
    g1[5] = (int)((unsigned)N_);                               // tensor_dim0_stride=4096
    g1[6] = 0;
    g1[7] = 0;
    const v4i z4 = {0, 0, 0, 0};                               // groups 2/3 unused (2D)
    const v8i z8 = {0, 0, 0, 0, 0, 0, 0, 0};
    __builtin_amdgcn_tensor_load_to_lds(g0, g1, z4, z4, z8, 0);
}

// ---------------------------------------------------------------------------
// Fused kernel-gen + row-normalized GEMM
// ---------------------------------------------------------------------------
__global__ __launch_bounds__(THREADS)
void csoc_rg_attention_kernel(const _Float16* __restrict__ latT,
                              const float* __restrict__ alpha,
                              float* __restrict__ out)
{
    __shared__ _Float16 sB[2][D_ * KTILE];    // double-buffered B tiles, [col][k]
    __shared__ _Float16 sA[2][2][16 * KTILE]; // double-buffered A frags per row-tile
    __shared__ float    sRS[2][16];           // final row sums per row-tile

    const int tid  = threadIdx.x;
    const int lane = tid & 31;
    const int wave = tid >> 5;       // 0..7
    const int rt   = wave >> 2;      // row-tile 0/1
    const int cg   = wave & 3;       // col-group 0..3 (128 cols each)
    const bool isBuilder = (cg == 0);   // waves 0,4 build A fragments
    const bool isDma     = (wave == 3); // wave 3 drives the TDM

    const int blocksPerBatch = N_ / MTILE;
    const int b  = blockIdx.x / blocksPerBatch;
    const int i0 = (blockIdx.x % blocksPerBatch) * MTILE;

    const int   rowInTile = lane & 15;
    const int   hiHalf    = lane >> 4;                 // 0: K-low half, 1: K-high half
    const int   i_glob    = i0 + rt * 16 + rowInTile;
    const float alpha_i   = alpha[b * N_ + i_glob];
    const float i_f       = (float)i_glob;

    const unsigned ldsOffB[2] = { (unsigned)(size_t)&sB[0][0],
                                  (unsigned)(size_t)&sB[1][0] };
    const size_t latBaseT = (size_t)b * D_ * N_;   // latT[b][d][n]
    const float C_EXP = 1.4426950408889634f / LAMBDA_RG;  // log2(e)/lambda

    v8f acc[8];
    #pragma unroll
    for (int t = 0; t < 8; ++t) acc[t] = (v8f){0,0,0,0,0,0,0,0};

    float rowsum = 0.0f;

    // A-tile builder: only cg==0 waves run this; writes fragment-layout LDS.
    // lanes 0-15: K {0..15}; lanes 16-31: K {16..31} (contiguous, matches B)
    auto buildA = [&](int j0, int abuf) {
        const float aj = alpha[b * N_ + j0 + lane];   // coalesced, 1 load/lane
        v16h af;
        #pragma unroll
        for (int h = 0; h < 16; ++h) {
            const int   kl  = h + hiHalf * 16;
            const float ap  = 0.5f * (alpha_i + __shfl(aj, kl, 32));
            const float d   = fabsf(i_f - (float)(j0 + kl)) + 1e-4f;
            // r^-ap * exp(-r/lambda) = exp2(-ap*log2(r) - r*log2e/lambda)
            const float val = exp2f(-ap * __log2f(d) - d * C_EXP);
            rowsum += val;
            af[h]   = (_Float16)val;
        }
        *(v16h*)&sA[abuf][rt][lane * 16] = af;
    };

    // ---- prologue: DMA tile 0, build A(0) ----
    if (isDma) tdm_load_tile(ldsOffB[0], latT + latBaseT);
    if (isBuilder) buildA(0, 0);
    if (isDma) __builtin_amdgcn_s_wait_tensorcnt(0);
    __syncthreads();

    for (int step = 0; step < NSTEPS; ++step) {
        const int buf = step & 1;

        // kick next tile's DMA first: overlaps the whole compute phase
        if (isDma && step + 1 < NSTEPS)
            tdm_load_tile(ldsOffB[buf ^ 1], latT + latBaseT + (step + 1) * KTILE);

        // everyone consumes the shared A fragment (2x ds_load_b128)
        const v16h afrag = *(const v16h*)&sA[buf][rt][lane * 16];

        // 8 WMMAs, B-fragments pipelined one tile ahead
        const _Float16* __restrict__ bbuf = &sB[buf][0];
        const int colBase = cg * 128 + rowInTile;
        v16h bf = *(const v16h*)(bbuf + colBase * KTILE + hiHalf * 16);
        #pragma unroll
        for (int t = 0; t < 8; ++t) {
            v16h bfn = bf;
            if (t < 7)
                bfn = *(const v16h*)(bbuf + (colBase + (t + 1) * 16) * KTILE + hiHalf * 16);
            acc[t] = __builtin_amdgcn_wmma_f32_16x16x32_f16(
                false, afrag, false, bf, (short)0, acc[t], false, false);
            bf = bfn;
        }

        // builders overlap next step's A generation with this step's WMMAs
        if (isBuilder && step + 1 < NSTEPS)
            buildA((step + 1) * KTILE, buf ^ 1);

        if (isDma) __builtin_amdgcn_s_wait_tensorcnt(0);  // next tile landed
        __syncthreads();   // phase boundary: buffers safe to swap
    }

    // ---- publish row sums (builders only) ----
    if (isBuilder) {
        rowsum += __shfl_xor(rowsum, 16, 32);  // combine K-halves of same row
        if (lane < 16) sRS[rt][lane] = rowsum;
    }
    __syncthreads();

    // ---- normalize rows and store f32 ----
    const size_t outBase = (size_t)b * N_ * D_;
    #pragma unroll
    for (int v = 0; v < 8; ++v) {
        const int   row   = v + hiHalf * 8;
        const float scale = 1.0f / (sRS[rt][row] + 1e-8f);
        const int   og    = i0 + rt * 16 + row;
        #pragma unroll
        for (int t = 0; t < 8; ++t) {
            const int col = cg * 128 + t * 16 + rowInTile;
            out[outBase + (size_t)og * D_ + col] = acc[t][v] * scale;
        }
    }
}

extern "C" void kernel_launch(void* const* d_in, const int* in_sizes, int n_in,
                              void* d_out, int out_size, void* d_ws, size_t ws_size,
                              hipStream_t stream) {
    (void)in_sizes; (void)n_in; (void)ws_size; (void)out_size;
    const float* latent = (const float*)d_in[0];
    const float* alpha  = (const float*)d_in[1];
    float*       out    = (float*)d_out;
    _Float16*    latT   = (_Float16*)d_ws;   // needs B_*D_*N_*2 = 16 MB of scratch

    dim3 gridT(N_ / 32, D_ / 32, B_);
    dim3 blockT(32, 8);
    hipLaunchKernelGGL(latent_cvt_transpose, gridT, blockT, 0, stream, latent, latT);

    dim3 grid(B_ * (N_ / MTILE));   // 512 workgroups
    dim3 block(THREADS);            // 8 wave32
    hipLaunchKernelGGL(csoc_rg_attention_kernel, grid, block, 0, stream,
                       latT, alpha, out);
}
